// EdgeMLPHead_10539849744622
// MI455X (gfx1250) — matrix operand order repaired and checked by reading, not compiled, not run on hardware
//
#include <hip/hip_runtime.h>

typedef __attribute__((ext_vector_type(16))) _Float16 v16h;
typedef __attribute__((ext_vector_type(8)))  _Float16 v8h;
typedef __attribute__((ext_vector_type(8)))  float    v8f;
typedef __attribute__((ext_vector_type(4)))  float    v4f;

#define EMB_DIM 128
#define IN_DIM  256
#define HIDDEN  64
#define WAVES_PER_BLOCK 8
#define BLOCK_THREADS (WAVES_PER_BLOCK * 32)

// LDS layout (in halves): W1^T [64][256] f16, then per-wave A-tile [16][256] f16
#define WT_HALFS    (HIDDEN * IN_DIM)   // 16384 halves = 32KB
#define ATILE_HALFS (16 * IN_DIM)       // 4096 halves  = 8KB per wave
#define LDS_HALFS   (WT_HALFS + WAVES_PER_BLOCK * ATILE_HALFS)  // 96KB total

__global__ __launch_bounds__(BLOCK_THREADS)
void edge_mlp_wmma(const float* __restrict__ emb,
                   const long long* __restrict__ eidx,
                   const float* __restrict__ W1,
                   const float* __restrict__ b1,
                   const float* __restrict__ W2,
                   const float* __restrict__ b2,
                   float* __restrict__ out,
                   int E, int nTiles) {
    extern __shared__ _Float16 lds[];
    _Float16* shWt = lds;
    const int tid  = threadIdx.x;
    const int wave = tid >> 5;
    const int lane = tid & 31;
    _Float16* shA = lds + WT_HALFS + wave * ATILE_HALFS;

    // ---- Stage W1^T into LDS as f16: shWt[n*256 + k] = (f16)W1[k*64 + n] ----
    for (int i = tid; i < WT_HALFS; i += BLOCK_THREADS) {
        int k = i >> 6, n = i & 63;
        shWt[n * IN_DIM + k] = (_Float16)W1[i];
    }
    __syncthreads();

    const int  col    = lane & 15;       // N column within a 16-wide subtile
    const bool hiHalf = (lane >= 16);

    // Epilogue constants: this lane's 4 hidden columns (col + 16t)
    float b1c[4], w2c[4];
    #pragma unroll
    for (int t = 0; t < 4; ++t) {
        b1c[t] = b1[col + 16 * t];
        w2c[t] = W2[col + 16 * t];
    }
    const float b2v = b2[0];

    const int waveGlobal = blockIdx.x * WAVES_PER_BLOCK + wave;
    const int waveStride = gridDim.x * WAVES_PER_BLOCK;

    for (int tile = waveGlobal; tile < nTiles; tile += waveStride) {
        const int eBase = tile * 16;

        // ---- Gather 16 edges' concat(src,tgt) embeddings -> LDS f16 tile ----
        // lanes 0..15 fetch src half (cols 0..127), lanes 16..31 tgt half (cols 128..255)
        const int sub = hiHalf ? (lane - 16) : lane;   // 8 floats per lane per row
        #pragma unroll 4
        for (int r = 0; r < 16; ++r) {
            int e = eBase + r; if (e >= E) e = E - 1;
            long long node = eidx[(hiHalf ? E : 0) + e];
            const float* src = emb + (size_t)node * EMB_DIM + sub * 8;
            v4f f0 = *(const v4f*)(src);
            v4f f1 = *(const v4f*)(src + 4);
            v8h h;
            #pragma unroll
            for (int j = 0; j < 4; ++j) { h[j] = (_Float16)f0[j]; h[4 + j] = (_Float16)f1[j]; }
            *(v8h*)(shA + r * IN_DIM + (hiHalf ? EMB_DIM : 0) + sub * 8) = h;
        }
        __builtin_amdgcn_wave_barrier();   // LDS ops in-order within wave; block reordering

        // ---- [16x256] @ [256x64] via v_wmma_f32_16x16x32_f16 ----
        v8f acc[4] = {v8f{}, v8f{}, v8f{}, v8f{}};
        const int arow = hiHalf ? (lane - 16) : lane;
        #pragma unroll
        for (int kb = 0; kb < IN_DIM; kb += 32) {
            // A fragment: lane<16 row=lane K=kb+{0..15};
            //             lane>=16 row=lane-16 K=kb+{16..31}
            const _Float16* ap = shA + arow * IN_DIM + kb + (hiHalf ? 16 : 0);
            v8h alo = *(const v8h*)(ap);
            v8h ahi = *(const v8h*)(ap + 8);
            v16h a = __builtin_shufflevector(alo, ahi,
                      0,1,2,3,4,5,6,7,8,9,10,11,12,13,14,15);
            #pragma unroll
            for (int t = 0; t < 4; ++t) {
                // B fragment: lane holds column (col+16t), K = kb + (hiHalf?16:0) + {0..15}
                const _Float16* bp = shWt + (col + 16 * t) * IN_DIM + kb + (hiHalf ? 16 : 0);
                v16h bfrag = *(const v16h*)(bp);   // 32 contiguous bytes in LDS
                acc[t] = __builtin_amdgcn_wmma_f32_16x16x32_f16(
                            false, a, false, bfrag, (short)0, acc[t], false, false);
            }
        }

        // ---- Fused bias + ReLU + (h @ W2) + b2, reduce across 16-lane groups ----
        // Accumulator layout: lane L, VGPR v -> C[M = v + (L<16?0:8), N = col + 16t]
        #pragma unroll
        for (int v = 0; v < 8; ++v) {
            float s = 0.0f;
            #pragma unroll
            for (int t = 0; t < 4; ++t) {
                float x = acc[t][v] + b1c[t];
                x = x > 0.0f ? x : 0.0f;
                s += x * w2c[t];
            }
            s += __shfl_xor(s, 1, 16);
            s += __shfl_xor(s, 2, 16);
            s += __shfl_xor(s, 4, 16);
            s += __shfl_xor(s, 8, 16);
            if (col == 0) {
                int e = eBase + v + (hiHalf ? 8 : 0);
                if (e < E) out[e] = s + b2v;
            }
        }
    }
}

extern "C" void kernel_launch(void* const* d_in, const int* in_sizes, int n_in,
                              void* d_out, int out_size, void* d_ws, size_t ws_size,
                              hipStream_t stream) {
    const float*     emb  = (const float*)d_in[0];
    const long long* eidx = (const long long*)d_in[1];   // int64 [2, E]
    const float*     W1   = (const float*)d_in[2];       // [256, 64]
    const float*     b1   = (const float*)d_in[3];       // [64]
    const float*     W2   = (const float*)d_in[4];       // [64, 1]
    const float*     b2   = (const float*)d_in[5];       // [1]
    float*           out  = (float*)d_out;

    const int E      = in_sizes[1] / 2;
    const int nTiles = (E + 15) / 16;
    int blocks = (nTiles + WAVES_PER_BLOCK - 1) / WAVES_PER_BLOCK;
    if (blocks > 1280) blocks = 1280;   // grid-stride: amortize W1 staging per block

    const size_t shmem = (size_t)LDS_HALFS * sizeof(_Float16);  // 96KB < 320KB/WGP
    hipLaunchKernelGGL(edge_mlp_wmma, dim3(blocks), dim3(BLOCK_THREADS), shmem, stream,
                       emb, eidx, W1, b1, W2, b2, out, E, nTiles);
}
